// Attention_40510131536383
// MI455X (gfx1250) — matrix operand order repaired and checked by reading, not compiled, not run on hardware
//
#include <hip/hip_runtime.h>

// ---------------------------------------------------------------------------
// MI455X (gfx1250, wave32) fused multi-head attention:
//   1) Q/K projections : bf16 WMMA GEMM (f32 in -> bf16 out, f32 accum),
//                        double-buffered LDS, one barrier per K-step
//      V projection    : same GEMM, epilogue writes V PRE-TRANSPOSED
//                        ([B][H][HDIM][SEQ]) so attention tiles are 2D DMA-able
//   2) flash attention : per-(b,h) online-softmax; K/V tiles streamed into a
//                        double-buffered LDS pipeline by the Tensor Data Mover
//                        (tensor_load_to_lds + s_wait_tensorcnt 2 semaphore),
//                        row reductions via v_permlane16
//   3) output proj     : bf16 WMMA GEMM (bf16 in -> f32 out)
// ---------------------------------------------------------------------------

typedef __attribute__((ext_vector_type(16))) __bf16 v16bf;
typedef __attribute__((ext_vector_type(8)))  __bf16 v8bf;
typedef __attribute__((ext_vector_type(8)))  float  v8f;
typedef __attribute__((ext_vector_type(4)))  float  v4f;
typedef __attribute__((ext_vector_type(4)))  unsigned int u32x4;
typedef __attribute__((ext_vector_type(8)))  int    i32x8;
typedef __attribute__((ext_vector_type(4)))  int    i32x4;

union bf16x16 { v16bf v; v8bf h[2]; };

#define BATCH   2
#define SEQ     2048
#define DMODEL  1024
#define HEADS   16
#define HDIM    64

#if __has_builtin(__builtin_amdgcn_tensor_load_to_lds) && \
    __has_builtin(__builtin_amdgcn_s_wait_tensorcnt)
#define USE_TDM 1
#endif

static __device__ __forceinline__ v8f wmma_bf16(v16bf a, v16bf b, v8f c) {
  // D = A(16x32 bf16) * B(32x16 bf16) + C(16x16 f32)
  return __builtin_amdgcn_wmma_f32_16x16x32_bf16(
      false, a, false, b, (short)0, c, false, false);
}

// ---- 16-lane-row butterfly reductions (v_permlane16_b32, pure VALU) -------
static __device__ __forceinline__ float pl16(float x, unsigned s0, unsigned s1) {
#if __has_builtin(__builtin_amdgcn_permlane16)
  unsigned u = __float_as_uint(x);
  return __uint_as_float(__builtin_amdgcn_permlane16(u, u, s0, s1, false, false));
#else
  (void)s0; (void)s1;
  return x;
#endif
}
static __device__ __forceinline__ float red16_max(float x) {
#if __has_builtin(__builtin_amdgcn_permlane16)
  x = fmaxf(x, pl16(x, 0x67452301u, 0xEFCDAB89u));  // xor 1
  x = fmaxf(x, pl16(x, 0x54761032u, 0xDCFE98BAu));  // xor 2
  x = fmaxf(x, pl16(x, 0x32107654u, 0xBA98FEDCu));  // xor 4
  x = fmaxf(x, pl16(x, 0xFEDCBA98u, 0x76543210u));  // xor 8
#else
#pragma unroll
  for (int m = 8; m >= 1; m >>= 1) x = fmaxf(x, __shfl_xor(x, m, 16));
#endif
  return x;
}
static __device__ __forceinline__ float red16_sum(float x) {
#if __has_builtin(__builtin_amdgcn_permlane16)
  x += pl16(x, 0x67452301u, 0xEFCDAB89u);
  x += pl16(x, 0x54761032u, 0xDCFE98BAu);
  x += pl16(x, 0x32107654u, 0xBA98FEDCu);
  x += pl16(x, 0xFEDCBA98u, 0x76543210u);
#else
#pragma unroll
  for (int m = 8; m >= 1; m >>= 1) x += __shfl_xor(x, m, 16);
#endif
  return x;
}

#ifdef USE_TDM
// ---- Tensor Data Mover: 2D bf16 tile, global -> LDS -----------------------
// D# per CDNA5 ISA: group0 = {count=1 | lds_addr | global_addr | type=2},
// group1 = {data_size=2B, tensor_dim0/1, tile_dim0/1, tensor_dim0_stride}.
static __device__ __forceinline__ void tdm_load_2d(
    unsigned lds_off, const void* gptr,
    unsigned tensor_d0, unsigned tensor_d1,
    unsigned tile_d0, unsigned tile_d1, unsigned stride0_elems)
{
  unsigned long long ga = (unsigned long long)(size_t)gptr;
  u32x4 g0;
  g0[0] = 1u;                                     // count=1 (valid user D#)
  g0[1] = lds_off;                                // LDS byte address
  g0[2] = (unsigned)(ga & 0xffffffffu);           // global_addr[31:0]
  g0[3] = (unsigned)((ga >> 32) & 0x01ffffffu)    // global_addr[56:32]
        | (2u << 30);                             // type = 2 ("image")
  i32x8 g1;
  g1[0] = (int)(1u << 16);                        // data_size = 1 -> 2 bytes
  g1[1] = (int)((tensor_d0 & 0xffffu) << 16);     // tensor_dim0[15:0] @bit48
  g1[2] = (int)((tensor_d0 >> 16) | ((tensor_d1 & 0xffffu) << 16));
  g1[3] = (int)((tensor_d1 >> 16) | (tile_d0 << 16));   // tile_dim0 @bit112
  g1[4] = (int)(tile_d1 & 0xffffu);               // tile_dim1; tile_dim2=0
  g1[5] = (int)stride0_elems;                     // tensor_dim0_stride[31:0]
  g1[6] = 0;
  g1[7] = 0;
  i32x4 z4 = {0, 0, 0, 0};
  i32x8 z8 = {0, 0, 0, 0, 0, 0, 0, 0};            // unused groups (2D tensor)
  __builtin_amdgcn_tensor_load_to_lds(g0, g1, z4, z4, z8, 0);
}
#endif

// 16-element row loader (overloaded on source precision) -> float staging
static __device__ __forceinline__ void load16(const float* p, float* tmp) {
#pragma unroll
  for (int j = 0; j < 4; ++j) {
    v4f x = *(const v4f*)(p + 4 * j);
    tmp[4 * j + 0] = x.x; tmp[4 * j + 1] = x.y;
    tmp[4 * j + 2] = x.z; tmp[4 * j + 3] = x.w;
  }
}
static __device__ __forceinline__ void load16(const __bf16* p, float* tmp) {
  v8bf x0 = *(const v8bf*)p;
  v8bf x1 = *(const v8bf*)(p + 8);
#pragma unroll
  for (int j = 0; j < 8; ++j) { tmp[j] = (float)x0[j]; tmp[8 + j] = (float)x1[j]; }
}
static __device__ __forceinline__ void cvt_store16(__bf16* dst, const float* tmp) {
  v8bf o0, o1;
#pragma unroll
  for (int j = 0; j < 8; ++j) { o0[j] = (__bf16)tmp[j]; o1[j] = (__bf16)tmp[8 + j]; }
  *(v8bf*)dst       = o0;
  *(v8bf*)(dst + 8) = o1;
}

// ---------------------------------------------------------------------------
// GEMM:  C[M,N] = A[M,K] @ W[N,K]^T + bias   (nn.Linear convention)
// Double-buffered LDS: stage K-step n+1 in registers while WMMAs consume
// step n; one barrier per step.  OMODE 0: row-major out.  OMODE 1: scatter
// into [B][H][HDIM][SEQ] (pre-transposed V for DMA-able attention tiles).
// ---------------------------------------------------------------------------
template <typename AT, typename OT, int OMODE>
__global__ __launch_bounds__(256) void gemm_bias_wmma(
    const AT* __restrict__ A, const float* __restrict__ W,
    const float* __restrict__ bias, OT* __restrict__ C,
    int M, int N, int K)
{
  __shared__ __bf16 Alds[2][128][32];   // 16 KB
  __shared__ __bf16 Wlds[2][128][32];   // 16 KB

  const int t  = threadIdx.x;
  const int w  = t >> 5;
  const int l  = t & 31;
  const int kh = l >> 4;        // K-half selector (fragment layouts)
  const int ln = l & 15;        // M (A) / N (B,C) lane coordinate
  const int m0 = blockIdx.y * 128;
  const int n0 = blockIdx.x * 128;

  v8f acc[8];
#pragma unroll
  for (int f = 0; f < 8; ++f)
#pragma unroll
    for (int i = 0; i < 8; ++i) acc[f][i] = 0.f;

  const int lr = t >> 1;            // tile loader: 2 threads/row, 16 elems ea
  const int lc = (t & 1) * 16;
  const AT*    arow = A + (size_t)(m0 + lr) * K + lc;
  const float* wrow = W + (size_t)(n0 + lr) * K + lc;

  float ta[16], tw[16];
  load16(arow, ta);                 // prologue: stage K-step 0 into buf 0
  load16(wrow, tw);
  cvt_store16(&Alds[0][lr][lc], ta);
  cvt_store16(&Wlds[0][lr][lc], tw);
  __syncthreads();

  for (int kk = 0; kk < K; kk += 32) {
    const int  buf  = (kk >> 5) & 1;
    const bool more = (kk + 32) < K;
    if (more) {                     // overlap next-step global loads w/ WMMA
      load16(arow + kk + 32, ta);
      load16(wrow + kk + 32, tw);
      if (kk + 64 < K) {            // speculative prefetch one step further
        __builtin_prefetch(arow + kk + 64, 0, 3);
        __builtin_prefetch(wrow + kk + 64, 0, 3);
      }
    }

    bf16x16 af;
    af.h[0] = *(const v8bf*)&Alds[buf][16 * w + ln][8 * kh];
    af.h[1] = *(const v8bf*)&Alds[buf][16 * w + ln][16 + 8 * kh];
#pragma unroll
    for (int f = 0; f < 8; ++f) {
      bf16x16 bfr;
      bfr.h[0] = *(const v8bf*)&Wlds[buf][16 * f + ln][8 * kh];
      bfr.h[1] = *(const v8bf*)&Wlds[buf][16 * f + ln][16 + 8 * kh];
      acc[f] = wmma_bf16(af.v, bfr.v, acc[f]);
    }

    if (more) {                     // write next step into the other buffer
      cvt_store16(&Alds[buf ^ 1][lr][lc], ta);
      cvt_store16(&Wlds[buf ^ 1][lr][lc], tw);
    }
    __syncthreads();                // single barrier per K-step
  }

#pragma unroll
  for (int f = 0; f < 8; ++f) {
    const int ncol = n0 + 16 * f + ln;
    const float bv = bias[ncol];
#pragma unroll
    for (int i = 0; i < 8; ++i) {
      const int mrow = m0 + 16 * w + i + 8 * kh;
      const float val = acc[f][i] + bv;
      if (OMODE == 1) {        // V: [B][H][HDIM][SEQ] pre-transposed scatter
        const int bb = mrow >> 11, ss = mrow & (SEQ - 1);
        const int hh = ncol >> 6, dd = ncol & (HDIM - 1);
        C[((size_t)((bb * HEADS + hh) * HDIM + dd) << 11) + ss] = (OT)val;
      } else {
        C[(size_t)mrow * N + ncol] = (OT)val;
      }
    }
  }
}

// ---------------------------------------------------------------------------
// Flash attention: one (b, h, 128-query-row) tile per 256-thread block.
// TDM pipeline: wave 0 issues the K/V DMAs for tile n+1 into the spare LDS
// buffer, then s_wait_tensorcnt(2) guarantees tile n's pair (in-order per
// wave) has landed before the publishing barrier. DMA latency is hidden
// behind the 8 WMMAs + online softmax of the current tile.
// ---------------------------------------------------------------------------
__global__ __launch_bounds__(256) void flash_attn(
    const __bf16* __restrict__ Q, const __bf16* __restrict__ K,
    const __bf16* __restrict__ Vt_ws, __bf16* __restrict__ O)
{
  __shared__ __bf16 Klds[2][32][64];    // 32 keys x 64 dim (row-major)   8 KB
  __shared__ __bf16 Vt[2][64][32];      // V^T tile [dim][key]            8 KB
  __shared__ __bf16 Plds[8][16][32];    // per-wave P re-layout scratch   8 KB

  const int t  = threadIdx.x;
  const int w  = t >> 5;
  const int l  = t & 31;
  const int kh = l >> 4;
  const int ln = l & 15;
  const int b  = blockIdx.z;
  const int h  = blockIdx.y;
  const int q0 = blockIdx.x * 128;

  // Q A-fragments (16 rows x 64 dim -> two 16x32 fragments), loaded once
  bf16x16 qa[2];
  {
    const __bf16* qrow =
        Q + (size_t)(b * SEQ + q0 + 16 * w + ln) * DMODEL + h * HDIM;
#pragma unroll
    for (int d = 0; d < 2; ++d) {
      qa[d].h[0] = *(const v8bf*)(qrow + 32 * d + 8 * kh);
      qa[d].h[1] = *(const v8bf*)(qrow + 32 * d + 16 + 8 * kh);
    }
  }

  float mrow[8], lrow[8];
  v8f oacc[4];
#pragma unroll
  for (int i = 0; i < 8; ++i) { mrow[i] = -1e30f; lrow[i] = 0.f; }
#pragma unroll
  for (int f = 0; f < 4; ++f)
#pragma unroll
    for (int i = 0; i < 8; ++i) oacc[f][i] = 0.f;

  const size_t kbase = (size_t)(b * SEQ) * DMODEL + h * HDIM;
  const size_t vbase = (size_t)((b * HEADS + h) * HDIM) << 11;   // * SEQ

#ifdef USE_TDM
  const unsigned klds_off = (unsigned)(size_t)&Klds[0][0][0];
  const unsigned vlds_off = (unsigned)(size_t)&Vt[0][0][0];
  if (w == 0) {   // prologue: DMA tile 0 into buffer 0 (TDM ignores EXEC)
    tdm_load_2d(klds_off, K + kbase, HDIM, 32, HDIM, 32, DMODEL);
    tdm_load_2d(vlds_off, Vt_ws + vbase, 32, HDIM, 32, HDIM, SEQ);
  }
#else
  const int lkey = t >> 3, ld0 = (t & 7) * 8;   // K tile loader coords
  const int vr   = t >> 2, vc  = (t & 3) * 8;   // V^T tile loader coords
#endif

  for (int it = 0; it < SEQ / 32; ++it) {
    const int buf = it & 1;
#ifdef USE_TDM
    if (w == 0) {
      if (it + 1 < SEQ / 32) {      // issue tile it+1 into the spare buffer
        const int kn = (it + 1) * 32;
        tdm_load_2d(klds_off + 4096u * (unsigned)(buf ^ 1),
                    K + kbase + (size_t)kn * DMODEL, HDIM, 32, HDIM, 32, DMODEL);
        tdm_load_2d(vlds_off + 4096u * (unsigned)(buf ^ 1),
                    Vt_ws + vbase + kn, 32, HDIM, 32, HDIM, SEQ);
        __builtin_amdgcn_s_wait_tensorcnt(2);   // tile it's pair has landed
      } else {
        __builtin_amdgcn_s_wait_tensorcnt(0);   // drain for the final tile
      }
    }
#else
    {
      const int k0 = it * 32;
      v8bf kv = *(const v8bf*)(K + kbase + (size_t)(k0 + lkey) * DMODEL + ld0);
      *(v8bf*)&Klds[buf][lkey][ld0] = kv;
      v8bf vv = *(const v8bf*)(Vt_ws + vbase + ((size_t)vr << 11) + k0 + vc);
      *(v8bf*)&Vt[buf][vr][vc] = vv;
    }
#endif
    __syncthreads();                // publish tile `it` to all waves

    // ---- S = Q @ K^T : two 16x16 score tiles (keys 0-15, 16-31) ----
    v8f s[2];
#pragma unroll
    for (int g = 0; g < 2; ++g) {
      v8f z;
#pragma unroll
      for (int i = 0; i < 8; ++i) z[i] = 0.f;
      const int krow = 16 * g + ln;
#pragma unroll
      for (int d = 0; d < 2; ++d) {
        bf16x16 bk;
        bk.h[0] = *(const v8bf*)&Klds[buf][krow][32 * d + 8 * kh];
        bk.h[1] = *(const v8bf*)&Klds[buf][krow][32 * d + 16 + 8 * kh];
        z = wmma_bf16(qa[d].v, bk.v, z);
      }
      s[g] = z;
    }

    // ---- online softmax; C-layout row m = i + 8*kh spans a 16-lane row ----
#pragma unroll
    for (int i = 0; i < 8; ++i) {
      const float sm   = red16_max(fmaxf(s[0][i], s[1][i]));
      const float mnew = fmaxf(mrow[i], sm);
      const float scl  = __expf(mrow[i] - mnew);
      const float p0   = __expf(s[0][i] - mnew);
      const float p1   = __expf(s[1][i] - mnew);
      const float rs   = red16_sum(p0 + p1);
      lrow[i] = lrow[i] * scl + rs;
      mrow[i] = mnew;
#pragma unroll
      for (int f = 0; f < 4; ++f) oacc[f][i] *= scl;
      const int pr = i + 8 * kh;        // wave-private; same-wave DS in-order
      Plds[w][pr][ln]      = (__bf16)p0;
      Plds[w][pr][16 + ln] = (__bf16)p1;
    }

    // ---- O += P @ V : P re-read in A-fragment layout, V^T tile as B ----
    bf16x16 pa;
    pa.h[0] = *(const v8bf*)&Plds[w][ln][8 * kh];
    pa.h[1] = *(const v8bf*)&Plds[w][ln][16 + 8 * kh];
#pragma unroll
    for (int f = 0; f < 4; ++f) {
      bf16x16 bv2;
      bv2.h[0] = *(const v8bf*)&Vt[buf][16 * f + ln][8 * kh];
      bv2.h[1] = *(const v8bf*)&Vt[buf][16 * f + ln][16 + 8 * kh];
      oacc[f] = wmma_bf16(pa.v, bv2.v, oacc[f]);
    }
    __syncthreads();                // all reads of buffer `buf` complete
  }

  // ---- normalize, write merged-head output (bf16 feed for final GEMM) ----
#pragma unroll
  for (int f = 0; f < 4; ++f) {
#pragma unroll
    for (int i = 0; i < 8; ++i) {
      const int mr  = q0 + 16 * w + i + 8 * kh;
      const int col = h * HDIM + 16 * f + ln;
      O[(size_t)(b * SEQ + mr) * DMODEL + col] = (__bf16)(oacc[f][i] / lrow[i]);
    }
  }
}

// ---------------------------------------------------------------------------
extern "C" void kernel_launch(void* const* d_in, const int* in_sizes, int n_in,
                              void* d_out, int out_size, void* d_ws, size_t ws_size,
                              hipStream_t stream) {
  (void)in_sizes; (void)n_in; (void)out_size; (void)ws_size;
  const float* q  = (const float*)d_in[0];
  const float* k  = (const float*)d_in[1];
  const float* v  = (const float*)d_in[2];
  const float* Wq = (const float*)d_in[3];
  const float* bq = (const float*)d_in[4];
  const float* Wk = (const float*)d_in[5];
  const float* bk = (const float*)d_in[6];
  const float* Wv = (const float*)d_in[7];
  const float* bv = (const float*)d_in[8];
  const float* Wo = (const float*)d_in[9];
  const float* bo = (const float*)d_in[10];
  float* out = (float*)d_out;

  const int M = BATCH * SEQ;            // 4096 flattened rows
  const size_t elems = (size_t)M * DMODEL;
  __bf16* qws = (__bf16*)d_ws;          // 8 MB each, 32 MB total
  __bf16* kws = qws + elems;
  __bf16* vws = kws + elems;            // pre-transposed V: [B][H][HDIM][SEQ]
  __bf16* ows = vws + elems;

  dim3 blk(256);
  dim3 gg(DMODEL / 128, M / 128);       // (8, 32)
  gemm_bias_wmma<float, __bf16, 0><<<gg, blk, 0, stream>>>(q, Wq, bq, qws, M, DMODEL, DMODEL);
  gemm_bias_wmma<float, __bf16, 0><<<gg, blk, 0, stream>>>(k, Wk, bk, kws, M, DMODEL, DMODEL);
  gemm_bias_wmma<float, __bf16, 1><<<gg, blk, 0, stream>>>(v, Wv, bv, vws, M, DMODEL, DMODEL);

  dim3 gf(SEQ / 128, HEADS, BATCH);     // (16, 16, 2)
  flash_attn<<<gf, blk, 0, stream>>>(qws, kws, vws, ows);

  gemm_bias_wmma<__bf16, float, 0><<<gg, blk, 0, stream>>>(ows, Wo, bo, out, M, DMODEL, DMODEL);
}
